// CategoryAttention_5016521802353
// MI455X (gfx1250) — matrix operand order repaired and checked by reading, not compiled, not run on hardware
//
#include <hip/hip_runtime.h>

// ---------------------------------------------------------------------------
// CategoryAttention for MI455X (gfx1250, wave32, WMMA f32_16x16x32_f16).
//
// Pipeline (all f16 storage in workspace, f32 accumulation in WMMA):
//   1) qkv_proj : Q = q@Wq^T+b (row-major f16), K likewise, V stored per-head
//                 transposed Vt[b][h][64][2048] f16.
//   2) attn     : streaming head-softmax attention. One workgroup = 16 waves
//                 = one 16-row q tile; wave h owns head h. Softmax is over
//                 the HEADS axis (faithful to the reference), so each k-chunk
//                 is fully normalizable on the spot -> no O(L^2) intermediate.
//   3) out_proj : AO @ Wo^T + b -> f32 d_out.
//
// Workspace: 4 * 4096*1024 f16 = 32 MB.
// ---------------------------------------------------------------------------

#define D_MODEL  1024
#define N_HEADS  16
#define HEAD_DIM 64
#define BATCH    2
#define SEQ      2048
#define BL       (BATCH * SEQ)   // 4096 rows

typedef __attribute__((ext_vector_type(16))) _Float16 v16h;
typedef __attribute__((ext_vector_type(8)))  _Float16 v8h;
typedef __attribute__((ext_vector_type(8)))  float    v8f;

#define WMMA_F16(a, b, c) \
  __builtin_amdgcn_wmma_f32_16x16x32_f16(false, (a), false, (b), (short)0, (c), false, false)

static __device__ __forceinline__ v16h join8(v8h lo, v8h hi) {
  v16h r;
#pragma unroll
  for (int i = 0; i < 8; ++i) { r[i] = lo[i]; r[i + 8] = hi[i]; }
  return r;
}

// load 8 consecutive f32 (32B aligned) and convert to 8 f16
static __device__ __forceinline__ v8h cvt8(const float* p) {
  float4 a = *(const float4*)p;
  float4 b = *(const float4*)(p + 4);
  v8h r;
  r[0] = (_Float16)a.x; r[1] = (_Float16)a.y; r[2] = (_Float16)a.z; r[3] = (_Float16)a.w;
  r[4] = (_Float16)b.x; r[5] = (_Float16)b.y; r[6] = (_Float16)b.z; r[7] = (_Float16)b.w;
  return r;
}

// A operand (16-bit 16x32): halves0-7 = K kb..kb+7, halves8-15 = K kb+16..kb+23 (kb = hi*8)
static __device__ __forceinline__ v16h ldA_f32(const float* row, int kb) {
  return join8(cvt8(row + kb), cvt8(row + kb + 16));
}
static __device__ __forceinline__ v16h ldA_f16(const _Float16* row, int kb) {
  return join8(*(const v8h*)(row + kb), *(const v8h*)(row + kb + 16));
}
// B operand (16-bit 32x16): halves0-7 = K kb/2..kb/2+7, halves8-15 = K kb/2+16..kb/2+23 (kb = hi*16)
static __device__ __forceinline__ v16h ldB_f32(const float* row, int kb) {
  return join8(cvt8(row + kb / 2), cvt8(row + kb / 2 + 16));
}
static __device__ __forceinline__ v16h ldB_f16(const _Float16* row, int kb) {
  return join8(*(const v8h*)(row + kb / 2), *(const v8h*)(row + kb / 2 + 16));
}

// ---------------------------------------------------------------------------
// Kernel 1: QKV projection.  grid = (1024, 3), block = 128 (4 waves).
// Each wave: one 16(m) x 64(n) output tile, K=1024 in 32-wide chunks.
// ---------------------------------------------------------------------------
__global__ __launch_bounds__(128) void qkv_proj_kernel(
    const float* __restrict__ q_in, const float* __restrict__ k_in,
    const float* __restrict__ v_in,
    const float* __restrict__ Wq, const float* __restrict__ bq,
    const float* __restrict__ Wk, const float* __restrict__ bk,
    const float* __restrict__ Wv, const float* __restrict__ bv,
    _Float16* __restrict__ Qws, _Float16* __restrict__ Kws,
    _Float16* __restrict__ Vtws)
{
  const int lane = threadIdx.x & 31;
  const int lm   = lane & 15;
  const int hi   = lane >> 4;
  const int tile = blockIdx.x * 4 + (threadIdx.x >> 5);   // 0..4095
  const int m0   = (tile >> 4) * 16;                      // input row tile
  const int n0   = (tile & 15) * 64;                      // output col tile
  const int which = blockIdx.y;                           // 0=Q 1=K 2=V

  const float* X    = which == 0 ? q_in : (which == 1 ? k_in : v_in);
  const float* W    = which == 0 ? Wq   : (which == 1 ? Wk   : Wv);
  const float* bias = which == 0 ? bq   : (which == 1 ? bk   : bv);

  v8f acc[4] = {};

  if (which < 2) {
    // Transposed orientation: D = W_tile(16n x 32d) x X^T(32d x 16m).
    // D lanes: N = m (fixed per lane), M = n (varies) -> contiguous row-major store.
    const float* xrow = X + (size_t)(m0 + lm) * D_MODEL;
    for (int kc = 0; kc < D_MODEL / 32; ++kc) {
      const int d0 = kc * 32;
      v16h bx = ldB_f32(xrow + d0, hi * 16);
#pragma unroll
      for (int s = 0; s < 4; ++s) {
        const float* wrow = W + (size_t)(n0 + s * 16 + lm) * D_MODEL;
        v16h aw = ldA_f32(wrow + d0, hi * 8);
        acc[s] = WMMA_F16(aw, bx, acc[s]);
      }
    }
    _Float16* OUT = (which == 0 ? Qws : Kws) + (size_t)(m0 + lm) * D_MODEL;
#pragma unroll
    for (int s = 0; s < 4; ++s) {
      const int nb = n0 + s * 16 + hi * 8;
      v8h o;
#pragma unroll
      for (int r = 0; r < 8; ++r) o[r] = (_Float16)(acc[s][r] + bias[nb + r]);
      *(v8h*)(OUT + nb) = o;
    }
  } else {
    // V: D = X_tile(16m x 32d) x W^T(32d x 16n).  N = n (fixed per lane),
    // M = m = sequence pos (varies) -> contiguous store into Vt[b][h][d][k].
    const float* xrow = X + (size_t)(m0 + lm) * D_MODEL;
    for (int kc = 0; kc < D_MODEL / 32; ++kc) {
      const int d0 = kc * 32;
      v16h ax = ldA_f32(xrow + d0, hi * 8);
#pragma unroll
      for (int s = 0; s < 4; ++s) {
        const float* wrow = W + (size_t)(n0 + s * 16 + lm) * D_MODEL;
        v16h bw = ldB_f32(wrow + d0, hi * 16);
        acc[s] = WMMA_F16(ax, bw, acc[s]);
      }
    }
    const int b     = m0 >> 11;                 // / SEQ
    const int kbase = (m0 & (SEQ - 1)) + hi * 8;
#pragma unroll
    for (int s = 0; s < 4; ++s) {
      const int   n  = n0 + s * 16 + lm;        // 0..1023
      const float bn = bias[n];
      v8h o;
#pragma unroll
      for (int r = 0; r < 8; ++r) o[r] = (_Float16)(acc[s][r] + bn);
      _Float16* dst = Vtws +
          ((size_t)((b * N_HEADS + (n >> 6)) * HEAD_DIM + (n & 63)) * SEQ) + kbase;
      *(v8h*)dst = o;
    }
  }
}

// ---------------------------------------------------------------------------
// Kernel 2: streaming attention with softmax over the HEADS axis.
// grid = B*(L/16) = 256 blocks, block = 512 threads (16 waves; wave h = head h).
// ---------------------------------------------------------------------------
__global__ __launch_bounds__(512) void attn_kernel(
    const _Float16* __restrict__ Qws, const _Float16* __restrict__ Kws,
    const _Float16* __restrict__ Vtws, _Float16* __restrict__ AOws)
{
  __shared__ __attribute__((aligned(16))) float    Elds[N_HEADS][16][33]; // f32 energies
  __shared__ __attribute__((aligned(16))) _Float16 Alds[N_HEADS][16][40]; // f16 attn

  const int tid  = threadIdx.x;
  const int lane = tid & 31, lm = lane & 15, hi = lane >> 4;
  const int h    = tid >> 5;                       // head / wave id
  const int b    = blockIdx.x >> 7;                // 128 q-tiles per batch
  const int q0   = (blockIdx.x & 127) * 16;

  // Persistent Q^T B-operands: B[k=hd][n=q], per lane one q row, contiguous hd.
  const _Float16* qrow = Qws + (size_t)(b * SEQ + q0 + lm) * D_MODEL + h * HEAD_DIM;
  const v16h bq0 = ldB_f16(qrow,      hi * 16);
  const v16h bq1 = ldB_f16(qrow + 32, hi * 16);

  v8f vo[4] = {};                                  // out^T accumulators (64d x 16q)

  const _Float16* Kbase = Kws  + (size_t)(b * SEQ) * D_MODEL + h * HEAD_DIM;
  const _Float16* Vbase = Vtws + (size_t)(b * N_HEADS + h) * HEAD_DIM * SEQ;

  const int sq = tid >> 5;     // softmax: this thread's q position (0..15)
  const int sk = tid & 31;     // softmax: this thread's k position (0..31)

  for (int kc = 0; kc < SEQ / 32; ++kc) {
    const int k0 = kc * 32;

    // ---- Phase 1: E^T(32k x 16q) = K_tile x Q^T, K-dim = 64 head dims ----
    v8f e0 = {}, e1 = {};
    {
      const _Float16* kr0 = Kbase + (size_t)(k0 + lm) * D_MODEL;
      const _Float16* kr1 = Kbase + (size_t)(k0 + 16 + lm) * D_MODEL;
      v16h a;
      a = ldA_f16(kr0,      hi * 8); e0 = WMMA_F16(a, bq0, e0);
      a = ldA_f16(kr0 + 32, hi * 8); e0 = WMMA_F16(a, bq1, e0);
      a = ldA_f16(kr1,      hi * 8); e1 = WMMA_F16(a, bq0, e1);
      a = ldA_f16(kr1 + 32, hi * 8); e1 = WMMA_F16(a, bq1, e1);
    }
    // D lanes: N = q = lm (fixed), M = k (varies) -> contiguous-k LDS stash.
#pragma unroll
    for (int r = 0; r < 8; ++r) {
      Elds[h][lm][hi * 8 + r]      = e0[r] * 0.125f;   // * 1/sqrt(64)
      Elds[h][lm][16 + hi * 8 + r] = e1[r] * 0.125f;
    }
    __syncthreads();

    // ---- Phase 2: softmax across the 16 heads at each (q,k) ----
    {
      float v[N_HEADS];
      float m = -3.0e38f;
#pragma unroll
      for (int hh = 0; hh < N_HEADS; ++hh) {
        v[hh] = Elds[hh][sq][sk];
        m = fmaxf(m, v[hh]);
      }
      float ssum = 0.0f;
#pragma unroll
      for (int hh = 0; hh < N_HEADS; ++hh) { v[hh] = __expf(v[hh] - m); ssum += v[hh]; }
      const float inv = __builtin_amdgcn_rcpf(ssum);
#pragma unroll
      for (int hh = 0; hh < N_HEADS; ++hh)
        Alds[hh][sq][sk] = (_Float16)(v[hh] * inv);
    }
    __syncthreads();

    // ---- Phase 3: out^T(64d x 16q) += Vt_chunk(64d x 32k) x attn^T(32k x 16q) ----
    const v16h ba = ldB_f16(&Alds[h][lm][0], hi * 16);
#pragma unroll
    for (int s = 0; s < 4; ++s) {
      const _Float16* vrow = Vbase + (size_t)(s * 16 + lm) * SEQ + k0;
      v16h av = ldA_f16(vrow, hi * 8);
      vo[s] = WMMA_F16(av, ba, vo[s]);
    }

    // Prefetch next K tile rows (global_prefetch_b8).
    if (kc + 1 < SEQ / 32)
      __builtin_prefetch(Kbase + (size_t)(k0 + 32 + lm) * D_MODEL, 0, 3);
  }

  // Store: out^T lanes have N = q = lm fixed, M = d consecutive -> v8h stores.
  _Float16* orow = AOws + (size_t)(b * SEQ + q0 + lm) * D_MODEL + h * HEAD_DIM;
#pragma unroll
  for (int s = 0; s < 4; ++s) {
    v8h o;
#pragma unroll
    for (int r = 0; r < 8; ++r) o[r] = (_Float16)vo[s][r];
    *(v8h*)(orow + s * 16 + hi * 8) = o;
  }
}

// ---------------------------------------------------------------------------
// Kernel 3: output projection AO @ Wo^T + b -> f32.  grid = 1024, block = 128.
// ---------------------------------------------------------------------------
__global__ __launch_bounds__(128) void out_proj_kernel(
    const _Float16* __restrict__ AOws, const float* __restrict__ Wo,
    const float* __restrict__ bo, float* __restrict__ out)
{
  const int lane = threadIdx.x & 31, lm = lane & 15, hi = lane >> 4;
  const int tile = blockIdx.x * 4 + (threadIdx.x >> 5);
  const int m0   = (tile >> 4) * 16;
  const int n0   = (tile & 15) * 64;

  v8f acc[4] = {};
  const _Float16* xrow = AOws + (size_t)(m0 + lm) * D_MODEL;
  for (int kc = 0; kc < D_MODEL / 32; ++kc) {
    const int d0 = kc * 32;
    v16h bx = ldB_f16(xrow + d0, hi * 16);
#pragma unroll
    for (int s = 0; s < 4; ++s) {
      const float* wrow = Wo + (size_t)(n0 + s * 16 + lm) * D_MODEL;
      v16h aw = ldA_f32(wrow + d0, hi * 8);
      acc[s] = WMMA_F16(aw, bx, acc[s]);
    }
  }
  float* orow = out + (size_t)(m0 + lm) * D_MODEL;
#pragma unroll
  for (int s = 0; s < 4; ++s) {
    const int nb = n0 + s * 16 + hi * 8;
    float4 o0, o1;
    o0.x = acc[s][0] + bo[nb + 0]; o0.y = acc[s][1] + bo[nb + 1];
    o0.z = acc[s][2] + bo[nb + 2]; o0.w = acc[s][3] + bo[nb + 3];
    o1.x = acc[s][4] + bo[nb + 4]; o1.y = acc[s][5] + bo[nb + 5];
    o1.z = acc[s][6] + bo[nb + 6]; o1.w = acc[s][7] + bo[nb + 7];
    *(float4*)(orow + nb)     = o0;
    *(float4*)(orow + nb + 4) = o1;
  }
}

// ---------------------------------------------------------------------------
extern "C" void kernel_launch(void* const* d_in, const int* in_sizes, int n_in,
                              void* d_out, int out_size, void* d_ws, size_t ws_size,
                              hipStream_t stream) {
  const float* q_in = (const float*)d_in[0];
  const float* k_in = (const float*)d_in[1];
  const float* v_in = (const float*)d_in[2];
  const float* Wq   = (const float*)d_in[3];
  const float* bq   = (const float*)d_in[4];
  const float* Wk   = (const float*)d_in[5];
  const float* bk   = (const float*)d_in[6];
  const float* Wv   = (const float*)d_in[7];
  const float* bv   = (const float*)d_in[8];
  const float* Wo   = (const float*)d_in[9];
  const float* bo   = (const float*)d_in[10];

  const size_t mat = (size_t)BL * D_MODEL;   // 4M f16 elements = 8 MB each
  _Float16* Qws = (_Float16*)d_ws;
  _Float16* Kws = Qws + mat;
  _Float16* Vt  = Kws + mat;
  _Float16* AO  = Vt  + mat;

  qkv_proj_kernel<<<dim3(1024, 3), 128, 0, stream>>>(
      q_in, k_in, v_in, Wq, bq, Wk, bk, Wv, bv, Qws, Kws, Vt);
  attn_kernel<<<dim3(256), 512, 0, stream>>>(Qws, Kws, Vt, AO);
  out_proj_kernel<<<dim3(1024), 128, 0, stream>>>(AO, Wo, bo, (float*)d_out);
}
